// CausalSelfAttention_34153579938125
// MI455X (gfx1250) — matrix operand order repaired and checked by reading, not compiled, not run on hardware
//
#include <hip/hip_runtime.h>

// CausalSelfAttention for MI455X (gfx1250, wave32, WMMA bf16, async global->LDS)
// B=4, T=2048, C=1024, NH=16, HD=64. fp32 in/out, bf16 WMMA compute, fp32 accum.

#define BATCH  4
#define SEQ    2048
#define CDIM   1024
#define NHEADS 16
#define HD     64

// GEMM block tiling: 128x128 output per 512-thread block (16 waves),
// K staged 64 at a time, each wave owns a 32x32 strip (4 accumulators).
#define BM 128
#define BN 128
#define BK 64

typedef __bf16 bf16_t;
typedef __attribute__((ext_vector_type(16))) __bf16 v16bf;
typedef __attribute__((ext_vector_type(8)))  float  v8f;
typedef int v4i __attribute__((vector_size(16)));   // matches async builtin's pointee type

union Frag { v16bf v; uint4 q[2]; };

// ---------------- CDNA5 async global->LDS copy (ASYNCcnt tracked) ----------------

__device__ __forceinline__ void async_copy_b128(const void* g, const void* lds_generic) {
  // Flat shared addresses are {shared_aperture_hi32, lds_offset32}; low 32 bits
  // are the LDS byte address.
  uint32_t lo = (uint32_t)(uintptr_t)lds_generic;
#if __has_builtin(__builtin_amdgcn_global_load_async_to_lds_b128)
  __builtin_amdgcn_global_load_async_to_lds_b128(
      (__attribute__((address_space(1))) v4i*)(uintptr_t)g,
      (__attribute__((address_space(3))) v4i*)(uintptr_t)lo, 0, 0);
#else
  asm volatile("global_load_async_to_lds_b128 %0, %1, off"
               :: "v"(lo), "v"(g) : "memory");
#endif
}

__device__ __forceinline__ void wait_async0() {
#if __has_builtin(__builtin_amdgcn_s_wait_asynccnt)
  __builtin_amdgcn_s_wait_asynccnt(0);
#else
  asm volatile("s_wait_asynccnt 0x0" ::: "memory");
#endif
}

// ---------------- WMMA fragment loaders ----------------

// A-matrix fragment (16x32 bf16, MxK): lanes 0-15 = rows, hold K{8h..8h+7} and K{16+8h..}
__device__ __forceinline__ v16bf load_fragA(const bf16_t* base, int row0, int kk, int stride) {
  const int lane = threadIdx.x & 31;
  const int hl = lane >> 4, lr = lane & 15;
  const char* p = (const char*)(base + (size_t)(row0 + lr) * stride + kk);
  Frag f;
  f.q[0] = *(const uint4*)(p + hl * 16);
  f.q[1] = *(const uint4*)(p + 32 + hl * 16);
  return f.v;
}

// B-matrix fragment (32x16 bf16, KxN) from an [N][K]-row-major tile:
// lane = column N, hold K{8h..8h+7} and K{16+8h..} (same order as A)
__device__ __forceinline__ v16bf load_fragB(const bf16_t* base, int n0, int kk, int stride) {
  const int lane = threadIdx.x & 31;
  const int hl = lane >> 4, lr = lane & 15;
  const char* p = (const char*)(base + (size_t)(n0 + lr) * stride + kk);
  Frag f;
  f.q[0] = *(const uint4*)(p + hl * 16);
  f.q[1] = *(const uint4*)(p + 32 + hl * 16);
  return f.v;
}

__device__ __forceinline__ v8f wmma_bf16(v16bf a, v16bf b, v8f c) {
  return __builtin_amdgcn_wmma_f32_16x16x32_bf16(false, a, false, b, (short)0, c, false, false);
}

// ---------------- conversion kernels ----------------

__global__ __launch_bounds__(256)
void cvt_f32_to_bf16_x4(const float* __restrict__ in, bf16_t* __restrict__ out, int n4) {
  int i = blockIdx.x * 256 + threadIdx.x;
  if (i < n4) {
    float4 f = ((const float4*)in)[i];
    out[4 * i + 0] = (bf16_t)f.x;
    out[4 * i + 1] = (bf16_t)f.y;
    out[4 * i + 2] = (bf16_t)f.z;
    out[4 * i + 3] = (bf16_t)f.w;
  }
}

// in: fp32 [K][N]  ->  out: bf16 [N][K]
__global__ __launch_bounds__(256)
void cvt_transpose_bf16(const float* __restrict__ in, bf16_t* __restrict__ out, int K, int N) {
  int o = blockIdx.x * 256 + threadIdx.x;
  if (o < K * N) {
    int nn = o / K, k = o - nn * K;
    out[o] = (bf16_t)in[(size_t)k * N + nn];
  }
}

// ---------------- shared GEMM core ----------------
// BMxBN tile of A[M][K] @ Wt[N][K]^T. Double-buffered LDS filled by async
// global->LDS b128 copies. 512 threads = 16 waves; wave w owns a 32x32 strip
// (acc[m 0..1][n 0..1], 32 VGPRs) -> low register pressure, no live-range
// splits. 8 WMMAs per single barrier per K-step. Threads 0-255 stage the A
// tile, threads 256-511 stage the B tile (4 async b128 copies each).

template <typename Epilogue>
__device__ __forceinline__ void gemm_core(const bf16_t* __restrict__ A,
                                          const bf16_t* __restrict__ Wt,
                                          int K, Epilogue epi) {
  __shared__ bf16_t As[2][BM * BK];
  __shared__ bf16_t Bs[2][BN * BK];
  const int tid = threadIdx.x;
  const int nbase = blockIdx.x * BN;
  const int mbase = blockIdx.y * BM;
  const int w = tid >> 5;
  const int wm = (w & 3) * 32;     // wave M offset: 2 subtiles of 16
  const int wn = (w >> 2) * 32;    // wave N offset: 2 subtiles of 16

  v8f acc[2][2] = {};

  // Per-thread staging: half the block stages A, half stages B; 4 b128 each.
  const int half = tid >> 8;       // 0 => A tile, 1 => B tile (uniform per wave)
  const int t2 = tid & 255;
  const bf16_t* gbase = half ? (Wt + (size_t)nbase * K) : (A + (size_t)mbase * K);
  const bf16_t* lbase = half ? &Bs[0][0] : &As[0][0];
  const bf16_t* gsrc[4];
  const bf16_t* ldst[4];
  #pragma unroll
  for (int i = 0; i < 4; ++i) {
    int idx = t2 + i * 256;                // 1024 uint4 per 128x64 bf16 tile
    int r = idx >> 3, c = (idx & 7) * 8;
    gsrc[i] = gbase + (size_t)r * K + c;
    ldst[i] = lbase + r * BK + c;
  }
  const size_t bufstep = (size_t)BM * BK;  // elements between buffer 0 and 1

  auto stage_tile = [&](int buf) {         // stages current gsrc, then advances by BK
    #pragma unroll
    for (int i = 0; i < 4; ++i) {
      async_copy_b128(gsrc[i], ldst[i] + (size_t)buf * bufstep);
      gsrc[i] += BK;
    }
  };

  stage_tile(0);
  wait_async0();
  __syncthreads();

  const int KT = K / BK;
  for (int kt = 0; kt < KT; ++kt) {
    const int cur = kt & 1;
    if (kt + 1 < KT) stage_tile(1 - cur);  // async copies overlap the WMMAs below
    #pragma unroll
    for (int kk = 0; kk < BK; kk += 32) {
      v16bf a0 = load_fragA(&As[cur][0], wm,      kk, BK);
      v16bf a1 = load_fragA(&As[cur][0], wm + 16, kk, BK);
      v16bf b0 = load_fragB(&Bs[cur][0], wn,      kk, BK);
      v16bf b1 = load_fragB(&Bs[cur][0], wn + 16, kk, BK);
      acc[0][0] = wmma_bf16(a0, b0, acc[0][0]);
      acc[1][0] = wmma_bf16(a1, b0, acc[1][0]);
      acc[0][1] = wmma_bf16(a0, b1, acc[0][1]);
      acc[1][1] = wmma_bf16(a1, b1, acc[1][1]);
    }
    if (kt + 1 < KT) wait_async0();        // our copies into 1-cur are done
    __syncthreads();                       // publish; single barrier per K-step
  }

  epi(acc, mbase, nbase, wm, wn);
}

// ---------------- QKV projection GEMM ----------------
// qkv = x @ w_attn + b_attn ; scatter into head-major bf16 q/k/v [B,NH,T,HD]

__global__ __launch_bounds__(512, 1)
void qkv_gemm_kernel(const bf16_t* __restrict__ A, const bf16_t* __restrict__ Wt,
                     const float* __restrict__ bias,
                     bf16_t* __restrict__ qh, bf16_t* __restrict__ kh, bf16_t* __restrict__ vh) {
  const int lane = threadIdx.x & 31, hl = lane >> 4, lr = lane & 15;
  gemm_core(A, Wt, CDIM,
    [&](const v8f (&acc)[2][2], int mbase, int nbase, int wm, int wn) {
      #pragma unroll
      for (int m = 0; m < 2; ++m) {
        #pragma unroll
        for (int n = 0; n < 2; ++n) {
          int ncol = nbase + wn + n * 16 + lr;
          float bv = bias[ncol];
          int part = ncol >> 10;       // 0=q, 1=k, 2=v
          int cc = ncol & 1023;
          int hh = cc >> 6, dd = cc & 63;
          bf16_t* dst = (part == 0) ? qh : (part == 1) ? kh : vh;
          #pragma unroll
          for (int r = 0; r < 8; ++r) {
            int mrow = mbase + wm + m * 16 + r + 8 * hl;   // global token row
            int bb = mrow >> 11, tt = mrow & 2047;
            dst[(((size_t)bb * NHEADS + hh) * SEQ + tt) * HD + dd] =
                (bf16_t)(acc[m][n][r] + bv);
          }
        }
      }
    });
}

// ---------------- output projection GEMM ----------------
// out = y @ w_proj + b_proj (fp32 output)

__global__ __launch_bounds__(512, 1)
void proj_gemm_kernel(const bf16_t* __restrict__ A, const bf16_t* __restrict__ Wt,
                      const float* __restrict__ bias, float* __restrict__ out) {
  const int lane = threadIdx.x & 31, hl = lane >> 4, lr = lane & 15;
  gemm_core(A, Wt, CDIM,
    [&](const v8f (&acc)[2][2], int mbase, int nbase, int wm, int wn) {
      #pragma unroll
      for (int m = 0; m < 2; ++m) {
        #pragma unroll
        for (int n = 0; n < 2; ++n) {
          int ncol = nbase + wn + n * 16 + lr;
          float bv = bias[ncol];
          #pragma unroll
          for (int r = 0; r < 8; ++r) {
            int mrow = mbase + wm + m * 16 + r + 8 * hl;
            out[(size_t)mrow * CDIM + ncol] = acc[m][n][r] + bv;
          }
        }
      }
    });
}

// ---------------- flash attention ----------------
// One block = 64 query rows of one (batch, head). 8 waves, O kept in registers.

__global__ __launch_bounds__(256)
void flash_attn_kernel(const bf16_t* __restrict__ qh, const bf16_t* __restrict__ kh,
                       const bf16_t* __restrict__ vh, bf16_t* __restrict__ yb) {
  __shared__ bf16_t Qs[64 * 64];   // [q][d]
  __shared__ bf16_t Ks[64 * 64];   // [key][d]
  __shared__ bf16_t Vt[64 * 64];   // [d][key] (transposed)
  __shared__ float  Sf[64 * 64];   // scores fp32
  __shared__ bf16_t Pb[64 * 64];   // probabilities bf16
  __shared__ float  mrow[64], lrow[64], arow[64];

  const int tid = threadIdx.x;
  const int q0 = blockIdx.x * 64;
  const int h = blockIdx.y, b = blockIdx.z;
  const size_t headoff = ((size_t)b * NHEADS + h) * SEQ * HD;
  const int w = tid >> 5, lane = tid & 31, hl = lane >> 4, lr = lane & 15;
  const int mi = w & 3;
  const int di0 = (w >> 2) * 2, di1 = di0 + 1;
  const int srow = tid >> 2, sseg = tid & 3;

  { // stage Q tile via async copies (no staging VGPRs)
    const bf16_t* src = qh + headoff + (size_t)(q0 + srow) * HD + sseg * 16;
    async_copy_b128(src,     Qs + srow * 64 + sseg * 16);
    async_copy_b128(src + 8, Qs + srow * 64 + sseg * 16 + 8);
  }
  if (tid < 64) { mrow[tid] = -3.0e38f; lrow[tid] = 0.f; }
  wait_async0();
  __syncthreads();

  v8f o0 = {}, o1 = {};
  const float scale = 0.125f;            // 1/sqrt(64)
  const int jmax = blockIdx.x;           // causal: key tiles 0..jmax

  for (int j = 0; j <= jmax; ++j) {
    { // stage K row-major (async), V transposed (b128 load + 16-bit scatter)
      const bf16_t* ksrc = kh + headoff + (size_t)(j * 64 + srow) * HD + sseg * 16;
      async_copy_b128(ksrc,     Ks + srow * 64 + sseg * 16);
      async_copy_b128(ksrc + 8, Ks + srow * 64 + sseg * 16 + 8);

      union { uint4 q; bf16_t e[8]; } va, vb;
      const uint4* vsrc = (const uint4*)(vh + headoff + (size_t)(j * 64 + srow) * HD + sseg * 16);
      va.q = vsrc[0]; vb.q = vsrc[1];
      #pragma unroll
      for (int e = 0; e < 8; ++e) Vt[(sseg * 16 + e) * 64 + srow] = va.e[e];
      #pragma unroll
      for (int e = 0; e < 8; ++e) Vt[(sseg * 16 + 8 + e) * 64 + srow] = vb.e[e];
    }
    if (j < jmax) {
      __builtin_prefetch(kh + headoff + (size_t)((j + 1) * 64) * HD, 0, 1);
      __builtin_prefetch(vh + headoff + (size_t)((j + 1) * 64) * HD, 0, 1);
    }
    wait_async0();
    __syncthreads();

    // S = Q K^T  (16x16 tiles; each wave does 2 of 16)
    #pragma unroll
    for (int t = 0; t < 2; ++t) {
      int ni = t ? di1 : di0;
      v8f s = {};
      #pragma unroll
      for (int kk = 0; kk < 64; kk += 32) {
        v16bf a  = load_fragA(Qs, mi * 16, kk, 64);
        v16bf bb = load_fragB(Ks, ni * 16, kk, 64);
        s = wmma_bf16(a, bb, s);
      }
      #pragma unroll
      for (int r = 0; r < 8; ++r)
        Sf[(mi * 16 + r + 8 * hl) * 64 + ni * 16 + lr] = s[r];
    }
    __syncthreads();

    // online softmax: one row per thread (threads 0..63)
    if (tid < 64) {
      const int row = tid;
      const int kmax = (q0 + row) - j * 64;   // columns <= kmax are causally valid
      float mo = mrow[row];
      float mx = mo;
      for (int c = 0; c < 64; ++c) {
        float sv = (c <= kmax) ? Sf[row * 64 + c] * scale : -3.0e38f;
        mx = fmaxf(mx, sv);
      }
      float alpha = __expf(mo - mx);
      float sum = 0.f;
      for (int c = 0; c < 64; ++c) {
        float p = (c <= kmax) ? __expf(Sf[row * 64 + c] * scale - mx) : 0.f;
        Pb[row * 64 + c] = (bf16_t)p;
        sum += p;
      }
      lrow[row] = lrow[row] * alpha + sum;
      mrow[row] = mx;
      arow[row] = alpha;
    }
    __syncthreads();

    // O = diag(alpha) * O + P @ V   (C fed into WMMA)
    #pragma unroll
    for (int t = 0; t < 2; ++t) {
      int di = t ? di1 : di0;
      v8f acc = t ? o1 : o0;
      #pragma unroll
      for (int r = 0; r < 8; ++r)
        acc[r] *= arow[mi * 16 + r + 8 * hl];
      #pragma unroll
      for (int kk = 0; kk < 64; kk += 32) {
        v16bf a  = load_fragA(Pb, mi * 16, kk, 64);
        v16bf bb = load_fragB(Vt, di * 16, kk, 64);
        acc = wmma_bf16(a, bb, acc);
      }
      if (t) o1 = acc; else o0 = acc;
    }
    __syncthreads();
  }

  // normalize and write y as bf16 in [B,T,C] layout for the proj GEMM
  #pragma unroll
  for (int t = 0; t < 2; ++t) {
    int di = t ? di1 : di0;
    v8f o = t ? o1 : o0;
    #pragma unroll
    for (int r = 0; r < 8; ++r) {
      int row = mi * 16 + r + 8 * hl;
      float v = o[r] / lrow[row];
      yb[((size_t)(b * SEQ + q0 + row)) * CDIM + h * HD + di * 16 + lr] = (bf16_t)v;
    }
  }
}

// ---------------- launcher ----------------

extern "C" void kernel_launch(void* const* d_in, const int* in_sizes, int n_in,
                              void* d_out, int out_size, void* d_ws, size_t ws_size,
                              hipStream_t stream) {
  (void)in_sizes; (void)n_in; (void)out_size; (void)ws_size;
  const float* x      = (const float*)d_in[0];
  const float* w_attn = (const float*)d_in[1];
  const float* b_attn = (const float*)d_in[2];
  const float* w_proj = (const float*)d_in[3];
  const float* b_proj = (const float*)d_in[4];
  float* out = (float*)d_out;

  const size_t nx = (size_t)BATCH * SEQ * CDIM;   // 8,388,608 elements
  char* ws = (char*)d_ws;
  size_t off = 0;
  auto carve = [&](size_t bytes) { void* p = ws + off; off += (bytes + 255) & ~(size_t)255; return p; };
  bf16_t* xb  = (bf16_t*)carve(nx * 2);                       // x in bf16
  bf16_t* waT = (bf16_t*)carve((size_t)3 * CDIM * CDIM * 2);  // w_attn^T bf16 [3C][C]
  bf16_t* wpT = (bf16_t*)carve((size_t)CDIM * CDIM * 2);      // w_proj^T bf16 [C][C]
  bf16_t* qhb = (bf16_t*)carve(nx * 2);                       // q [B,NH,T,HD]
  bf16_t* khb = (bf16_t*)carve(nx * 2);                       // k [B,NH,T,HD]
  bf16_t* vhb = (bf16_t*)carve(nx * 2);                       // v [B,NH,T,HD]
  bf16_t* yb  = (bf16_t*)carve(nx * 2);                       // attn output [B,T,C]

  cvt_f32_to_bf16_x4<<<(int)(nx / 1024), 256, 0, stream>>>(x, xb, (int)(nx / 4));
  cvt_transpose_bf16<<<(3 * CDIM * CDIM + 255) / 256, 256, 0, stream>>>(w_attn, waT, CDIM, 3 * CDIM);
  cvt_transpose_bf16<<<(CDIM * CDIM + 255) / 256, 256, 0, stream>>>(w_proj, wpT, CDIM, CDIM);

  qkv_gemm_kernel<<<dim3(3 * CDIM / BN, BATCH * SEQ / BM), 512, 0, stream>>>(
      xb, waT, b_attn, qhb, khb, vhb);

  flash_attn_kernel<<<dim3(SEQ / 64, NHEADS, BATCH), 256, 0, stream>>>(qhb, khb, vhb, yb);

  proj_gemm_kernel<<<dim3(CDIM / BN, BATCH * SEQ / BM), 512, 0, stream>>>(yb, wpT, b_proj, out);
}
